// CrossAttention_13692355739986
// MI455X (gfx1250) — matrix operand order repaired and checked by reading, not compiled, not run on hardware
//
#include <hip/hip_runtime.h>

// ---------------------------------------------------------------------------
// CDNA5 / gfx1250 flash-attention pipeline in bf16 WMMA (fp32 accumulate).
//   B=4, N=2048, D=512, H=8, Dh=64, SCALE = 1/8
// ---------------------------------------------------------------------------

typedef __attribute__((ext_vector_type(16))) __bf16       v16bf;
typedef __attribute__((ext_vector_type(8)))  float        v8f;
typedef __attribute__((ext_vector_type(8)))  unsigned int v8u;

#define WMMA_BF16(a, b, c) \
  __builtin_amdgcn_wmma_f32_16x16x32_bf16(false, (a), false, (b), (short)0, (c), false, false)

static __device__ __forceinline__ unsigned short f2bf(float f) {
  unsigned u = __builtin_bit_cast(unsigned, f);
  u += 0x7fffu + ((u >> 16) & 1u);          // round-to-nearest-even
  return (unsigned short)(u >> 16);
}

// A-fragment (16xK tile, 16-bit): row M = lane&15; element e -> K = e + (e&8) + 8*half
// Source is row-major with leading dim `ld` (elements); pairs (2i,2i+1) are contiguous.
static __device__ __forceinline__ v16bf load_a_frag(const unsigned short* base, int ld, int lane) {
  const int m = lane & 15, h = (lane >> 4) & 1;
  const unsigned short* row = base + m * ld;
  v8u t;
#pragma unroll
  for (int i = 0; i < 8; ++i) {
    const int e = 2 * i;
    const int k = e + (e & 8) + 8 * h;
    t[i] = *(const unsigned*)(row + k);
  }
  return __builtin_bit_cast(v16bf, t);
}

// B-fragment (Kx16 tile, 16-bit): col N = lane&15; element e -> K = e + (e&8) + 8*half
// Source laid out so that for fixed N the K axis is contiguous (stride `ld` between N's).
static __device__ __forceinline__ v16bf load_b_frag(const unsigned short* base, int ld, int lane) {
  const int n = lane & 15, h = (lane >> 4) & 1;
  const unsigned short* col = base + n * ld;
  v8u t;
#pragma unroll
  for (int i = 0; i < 8; ++i) {
    const int k = 2 * i + ((2 * i) & 8) + 8 * h;
    t[i] = *(const unsigned*)(col + k);
  }
  return __builtin_bit_cast(v16bf, t);
}

// ---------------------------------------------------------------------------
// fp32 -> bf16 conversion
// ---------------------------------------------------------------------------
__global__ __launch_bounds__(256) void cvt_f32_bf16(const float* __restrict__ s,
                                                    unsigned short* __restrict__ d, int n) {
  int i = blockIdx.x * blockDim.x + threadIdx.x;
  if (i < n) d[i] = f2bf(s[i]);
}

// ---------------------------------------------------------------------------
// Q projection: q = x @ Wq^T  (M=8192, N=512, K=512), output bf16 [B,H,N,64]
// One 16x16 output tile per wave, 8 waves per block.
// ---------------------------------------------------------------------------
__global__ __launch_bounds__(256) void qproj_kernel(const unsigned short* __restrict__ xb,   // [8192][512]
                                                    const unsigned short* __restrict__ wq,   // [512][512]
                                                    unsigned short* __restrict__ qbf) {      // [4][8][2048][64]
  const int lane = threadIdx.x & 31;
  const int wave = threadIdx.x >> 5;
  const int tile = blockIdx.x * 8 + wave;
  const int mt = tile >> 5;   // 512 row tiles
  const int nt = tile & 31;   // 32 col tiles
  const unsigned short* abase = xb + mt * 16 * 512;
  const unsigned short* bbase = wq + nt * 16 * 512;   // W[n][k]: K contiguous per n
  v8f c = {};
  for (int k = 0; k < 512; k += 32) {
    __builtin_prefetch(abase + k + 64, 0, 1);
    v16bf a = load_a_frag(abase + k, 512, lane);
    v16bf b = load_b_frag(bbase + k, 512, lane);
    c = WMMA_BF16(a, b, c);
  }
  const int h = (lane >> 4) & 1;
  const int col = nt * 16 + (lane & 15);
  const int head = col >> 6, dh = col & 63;
#pragma unroll
  for (int r = 0; r < 8; ++r) {
    const int g = mt * 16 + r + 8 * h;          // global row
    const int b_ = g >> 11, nrow = g & 2047;
    qbf[(((b_ * 8 + head) * 2048) + nrow) * 64 + dh] = f2bf(c[r]);
  }
}

// ---------------------------------------------------------------------------
// Flash attention: q=k=v. One block per (b,h, 128-row band); wave owns 16 rows.
// ---------------------------------------------------------------------------
__global__ __launch_bounds__(256) void attn_kernel(const unsigned short* __restrict__ qbf,  // [32][2048][64]
                                                   unsigned short* __restrict__ obf) {      // [8192][512]
  __shared__ __align__(16) unsigned short Klds[32 * 64];  // [key][dim]  row-major
  __shared__ __align__(16) unsigned short Ktls[64 * 32];  // [dim][key]  transposed
  __shared__ __align__(16) unsigned short Plds[8 * 16 * 32]; // per-wave P scratch

  const int tid = threadIdx.x;
  const int lane = tid & 31;
  const int wave = tid >> 5;
  const int bh = blockIdx.x >> 4;   // 0..31  (b*8 + head)
  const int rb = blockIdx.x & 15;   // 128-row band
  const unsigned short* qh = qbf + bh * 2048 * 64;
  const int row0 = rb * 128 + wave * 16;

  // Q fragments for this wave's 16 rows (d 0..31 and 32..63), resident all loop.
  const v16bf qa0 = load_a_frag(qh + row0 * 64 + 0, 64, lane);
  const v16bf qa1 = load_a_frag(qh + row0 * 64 + 32, 64, lane);

  v8f o0 = {}, o1 = {}, o2 = {}, o3 = {};
  float mrow[8], lrow[8];
#pragma unroll
  for (int r = 0; r < 8; ++r) { mrow[r] = -3.0e38f; lrow[r] = 0.0f; }

  for (int j = 0; j < 2048; j += 32) {
    __syncthreads();
    {   // cooperative stage of 32-key tile (dual layout), 16B per thread
      const int key = tid >> 3;
      const int d0 = (tid & 7) * 8;
      const uint4 v = *(const uint4*)(qh + (j + key) * 64 + d0);
      *(uint4*)(&Klds[key * 64 + d0]) = v;
      const unsigned w0 = v.x, w1 = v.y, w2 = v.z, w3 = v.w;
      Ktls[(d0 + 0) * 32 + key] = (unsigned short)w0;
      Ktls[(d0 + 1) * 32 + key] = (unsigned short)(w0 >> 16);
      Ktls[(d0 + 2) * 32 + key] = (unsigned short)w1;
      Ktls[(d0 + 3) * 32 + key] = (unsigned short)(w1 >> 16);
      Ktls[(d0 + 4) * 32 + key] = (unsigned short)w2;
      Ktls[(d0 + 5) * 32 + key] = (unsigned short)(w2 >> 16);
      Ktls[(d0 + 6) * 32 + key] = (unsigned short)w3;
      Ktls[(d0 + 7) * 32 + key] = (unsigned short)(w3 >> 16);
    }
    __syncthreads();

    // S = Q . K^T : two 16x16 tiles (keys j..j+15 and j+16..j+31)
    v16bf b00 = load_b_frag(&Klds[0 * 64 + 0], 64, lane);
    v16bf b01 = load_b_frag(&Klds[0 * 64 + 32], 64, lane);
    v16bf b10 = load_b_frag(&Klds[16 * 64 + 0], 64, lane);
    v16bf b11 = load_b_frag(&Klds[16 * 64 + 32], 64, lane);
    v8f s0 = {}, s1 = {};
    s0 = WMMA_BF16(qa0, b00, s0);
    s0 = WMMA_BF16(qa1, b01, s0);
    s1 = WMMA_BF16(qa0, b10, s1);
    s1 = WMMA_BF16(qa1, b11, s1);

    // online softmax: rows live at (reg r + 8*half), cols spread over 16 lanes
    float al[8], p0[8], p1[8];
#pragma unroll
    for (int r = 0; r < 8; ++r) {
      const float x0 = s0[r] * 0.125f;
      const float x1 = s1[r] * 0.125f;
      float v = fmaxf(x0, x1);
      v = fmaxf(v, __shfl_xor(v, 1));
      v = fmaxf(v, __shfl_xor(v, 2));
      v = fmaxf(v, __shfl_xor(v, 4));
      v = fmaxf(v, __shfl_xor(v, 8));
      const float mn = fmaxf(mrow[r], v);
      const float a = __expf(mrow[r] - mn);
      mrow[r] = mn;
      const float e0 = __expf(x0 - mn);
      const float e1 = __expf(x1 - mn);
      float ps = e0 + e1;
      ps += __shfl_xor(ps, 1);
      ps += __shfl_xor(ps, 2);
      ps += __shfl_xor(ps, 4);
      ps += __shfl_xor(ps, 8);
      lrow[r] = lrow[r] * a + ps;
      al[r] = a; p0[r] = e0; p1[r] = e1;
    }
#pragma unroll
    for (int r = 0; r < 8; ++r) {
      o0[r] *= al[r]; o1[r] *= al[r]; o2[r] *= al[r]; o3[r] *= al[r];
    }

    // P (C layout) -> per-wave LDS scratch -> A-fragment layout
    {
      const int h = (lane >> 4) & 1, c = lane & 15;
      unsigned short* pw = &Plds[wave * 512];
#pragma unroll
      for (int r = 0; r < 8; ++r) {
        const int row = r + 8 * h;
        pw[row * 32 + c]      = f2bf(p0[r]);
        pw[row * 32 + 16 + c] = f2bf(p1[r]);
      }
    }
    v16bf pa  = load_a_frag(&Plds[wave * 512], 32, lane);   // LDS in-order per wave
    v16bf vb0 = load_b_frag(&Ktls[0 * 16 * 32], 32, lane);  // V = K tile, [dim][key]
    v16bf vb1 = load_b_frag(&Ktls[1 * 16 * 32], 32, lane);
    v16bf vb2 = load_b_frag(&Ktls[2 * 16 * 32], 32, lane);
    v16bf vb3 = load_b_frag(&Ktls[3 * 16 * 32], 32, lane);
    o0 = WMMA_BF16(pa, vb0, o0);
    o1 = WMMA_BF16(pa, vb1, o1);
    o2 = WMMA_BF16(pa, vb2, o2);
    o3 = WMMA_BF16(pa, vb3, o3);
  }

  // normalize + scatter back to [B,N,512] bf16
  const int h = (lane >> 4) & 1, c = lane & 15;
  const int b_ = bh >> 3, head = bh & 7;
#pragma unroll
  for (int r = 0; r < 8; ++r) {
    const float inv = 1.0f / lrow[r];
    const int row = row0 + r + 8 * h;
    unsigned short* orow = obf + (b_ * 2048 + row) * 512 + head * 64 + c;
    orow[0]  = f2bf(o0[r] * inv);
    orow[16] = f2bf(o1[r] * inv);
    orow[32] = f2bf(o2[r] * inv);
    orow[48] = f2bf(o3[r] * inv);
  }
}

// ---------------------------------------------------------------------------
// Output projection: out = o @ Wo^T + bo  (fp32 output)
// ---------------------------------------------------------------------------
__global__ __launch_bounds__(256) void oproj_kernel(const unsigned short* __restrict__ ob,  // [8192][512]
                                                    const unsigned short* __restrict__ wo,  // [512][512]
                                                    const float* __restrict__ bo,
                                                    float* __restrict__ out) {              // [8192][512]
  const int lane = threadIdx.x & 31;
  const int wave = threadIdx.x >> 5;
  const int tile = blockIdx.x * 8 + wave;
  const int mt = tile >> 5;
  const int nt = tile & 31;
  const unsigned short* abase = ob + mt * 16 * 512;
  const unsigned short* bbase = wo + nt * 16 * 512;
  v8f c = {};
  for (int k = 0; k < 512; k += 32) {
    __builtin_prefetch(abase + k + 64, 0, 1);
    v16bf a = load_a_frag(abase + k, 512, lane);
    v16bf b = load_b_frag(bbase + k, 512, lane);
    c = WMMA_BF16(a, b, c);
  }
  const int h = (lane >> 4) & 1;
  const int col = nt * 16 + (lane & 15);
  const float bias = bo[col];
#pragma unroll
  for (int r = 0; r < 8; ++r) {
    const int g = mt * 16 + r + 8 * h;
    out[g * 512 + col] = c[r] + bias;
  }
}

// ---------------------------------------------------------------------------
// Launch. Inputs: x[4,2048,512] f32, Wq[512,512] f32, Wo[512,512] f32, bo[512] f32
// Workspace layout (bytes):
//   [0,       8M)  x_bf   (8192x512 bf16)   -- reused as attention output o_bf
//   [8M,    8.5M)  Wq_bf  (512x512 bf16)
//   [8.5M,    9M)  Wo_bf  (512x512 bf16)
//   [9M,     17M)  q_bf   ([4][8][2048][64] bf16)
// ---------------------------------------------------------------------------
extern "C" void kernel_launch(void* const* d_in, const int* in_sizes, int n_in,
                              void* d_out, int out_size, void* d_ws, size_t ws_size,
                              hipStream_t stream) {
  const float* x  = (const float*)d_in[0];
  const float* Wq = (const float*)d_in[1];
  const float* Wo = (const float*)d_in[2];
  const float* bo = (const float*)d_in[3];
  float* out = (float*)d_out;

  char* ws = (char*)d_ws;
  unsigned short* xb  = (unsigned short*)(ws);
  unsigned short* wqb = (unsigned short*)(ws + (8u << 20));
  unsigned short* wob = (unsigned short*)(ws + (8u << 20) + (512u << 10));
  unsigned short* qbf = (unsigned short*)(ws + (9u << 20));
  unsigned short* obf = xb;   // x_bf is dead after qproj; reuse for attention output

  const int nx = 4 * 2048 * 512;
  const int nw = 512 * 512;
  cvt_f32_bf16<<<(nx + 255) / 256, 256, 0, stream>>>(x, xb, nx);
  cvt_f32_bf16<<<(nw + 255) / 256, 256, 0, stream>>>(Wq, wqb, nw);
  cvt_f32_bf16<<<(nw + 255) / 256, 256, 0, stream>>>(Wo, wob, nw);

  // 8192/16 * 512/16 = 16384 tiles, 8 tiles (waves) per 256-thread block
  qproj_kernel<<<2048, 256, 0, stream>>>(xb, wqb, qbf);

  // 32 (b,h) * 16 row-bands of 128
  attn_kernel<<<512, 256, 0, stream>>>(qbf, obf);

  oproj_kernel<<<2048, 256, 0, stream>>>(obf, wob, bo, out);
}
